// Block_8512625181077
// MI455X (gfx1250) — compile-verified, matrix operand order checked by reading, not run on hardware
//
#include <hip/hip_runtime.h>
#include <hip/hip_bf16.h>
#include <math.h>

// ---------------- shapes ----------------
#define NB    16
#define CDIM  128
#define HH    56
#define WW    56
#define LL    (HH*WW)      // 3136
#define HDc   64
#define NHEAD 8
#define DSTn  16
#define DIc   128
#define DTRn  4
#define KDIR  4
#define EPSV  1e-5f

typedef __attribute__((ext_vector_type(16))) _Float16 v16h;
typedef __attribute__((ext_vector_type(8)))  _Float16 v8h;
typedef __attribute__((ext_vector_type(8)))  float    v8f;
typedef __attribute__((ext_vector_type(4)))  float    v4f;

// ---------------- WMMA helpers (v_wmma_f32_16x16x32_f16) ----------------
// 16-bit A layout: lane = M row (lane&15), halves 0..7 -> K = kb..kb+7,
// halves 8..15 -> K = kb+16..kb+23, kb = (lane>=16)*8.  Two b128 LDS loads.
__device__ __forceinline__ v16h frag_A_lds(const _Float16* src, int ld, int row0,
                                           int col0, int lane) {
  int m = row0 + (lane & 15);
  int kb = col0 + ((lane >= 16) ? 8 : 0);
  const _Float16* p = src + m * ld + kb;
  v8h lo = *(const v8h*)p;
  v8h hi = *(const v8h*)(p + 16);
  return __builtin_shufflevector(lo, hi, 0, 1, 2, 3, 4, 5, 6, 7,
                                 8, 9, 10, 11, 12, 13, 14, 15);
}

// B fragment for C = A * W^T, W row-major [N][K] f32 global. No bounds; caller
// guarantees n0+15 < N. Four global b128 loads.
__device__ __forceinline__ v16h frag_B_WT(const float* W, int ldw, int n0, int k0,
                                          int lane) {
  int n = n0 + (lane & 15);
  int kb = k0 + ((lane >= 16) ? 8 : 0);
  const float* p = W + (size_t)n * ldw + kb;
  v4f a0 = *(const v4f*)(p);
  v4f a1 = *(const v4f*)(p + 4);
  v4f b0 = *(const v4f*)(p + 16);
  v4f b1 = *(const v4f*)(p + 20);
  v8h lo, hi;
#pragma unroll
  for (int h = 0; h < 4; ++h) {
    lo[h] = (_Float16)a0[h]; lo[h + 4] = (_Float16)a1[h];
    hi[h] = (_Float16)b0[h]; hi[h + 4] = (_Float16)b1[h];
  }
  return __builtin_shufflevector(lo, hi, 0, 1, 2, 3, 4, 5, 6, 7,
                                 8, 9, 10, 11, 12, 13, 14, 15);
}

// Same with per-K gain folded in (RMS gains).
__device__ __forceinline__ v16h frag_B_WT_g(const float* W, int ldw, int n0, int k0,
                                            int lane, const float* gain) {
  int n = n0 + (lane & 15);
  int kb = k0 + ((lane >= 16) ? 8 : 0);
  const float* p = W + (size_t)n * ldw + kb;
  v4f a0 = *(const v4f*)(p);
  v4f a1 = *(const v4f*)(p + 4);
  v4f b0 = *(const v4f*)(p + 16);
  v4f b1 = *(const v4f*)(p + 20);
  v4f g0 = *(const v4f*)(gain + kb);
  v4f g1 = *(const v4f*)(gain + kb + 4);
  v4f g2 = *(const v4f*)(gain + kb + 16);
  v4f g3 = *(const v4f*)(gain + kb + 20);
  v8h lo, hi;
#pragma unroll
  for (int h = 0; h < 4; ++h) {
    lo[h] = (_Float16)(a0[h] * g0[h]); lo[h + 4] = (_Float16)(a1[h] * g1[h]);
    hi[h] = (_Float16)(b0[h] * g2[h]); hi[h + 4] = (_Float16)(b1[h] * g3[h]);
  }
  return __builtin_shufflevector(lo, hi, 0, 1, 2, 3, 4, 5, 6, 7,
                                 8, 9, 10, 11, 12, 13, 14, 15);
}

__device__ __forceinline__ v8f wmma_f16(v16h a, v16h b, v8f c) {
  return __builtin_amdgcn_wmma_f32_16x16x32_f16(false, a, false, b, (short)0, c,
                                                false, false);
}

__device__ __forceinline__ float gelu_t(float x) {
  float x3 = x * x * x;
  return 0.5f * x * (1.0f + tanhf(0.7978845608f * (x + 0.044715f * x3)));
}
__device__ __forceinline__ float sigm(float x) { return 1.0f / (1.0f + expf(-x)); }

// cross_scan source index
__device__ __forceinline__ int scan_src(int k, int lp) {
  int l = (k >= 2) ? (LL - 1 - lp) : lp;
  if (k & 1) l = (l % WW) * WW + (l / WW);
  return l;
}

// ======================= 1) window attention =======================
__global__ __launch_bounds__(256)
void k_attn(const float* __restrict__ x, const float* __restrict__ wq,
            const float* __restrict__ wk, const float* __restrict__ wv,
            const float* __restrict__ wproj, const float* __restrict__ bproj,
            const float* __restrict__ gq, const float* __restrict__ gk,
            const float* __restrict__ gv, float* __restrict__ y) {
  __shared__ __align__(16) _Float16 xn[64 * 64];   // padded 49 -> 64 rows
  __shared__ __align__(16) _Float16 qs[64 * 64];
  __shared__ __align__(16) _Float16 ksm[64 * 64];
  __shared__ __align__(16) _Float16 vsm[64 * 64];
  __shared__ __align__(16) _Float16 osm[64 * 64];
  int tid = threadIdx.x, lane = tid & 31, wave = tid >> 5;
  int wi = blockIdx.x;
  int b = wi >> 6, wh = (wi >> 3) & 7, ww = wi & 7;

  // per-token RMS into padded LDS (pad rows zeroed -> no bounds in fragments)
  for (int t = tid; t < 64; t += 256) {
    if (t < 49) {
      int hg = wh * 7 + t / 7, wg = ww * 7 + t % 7;
      const float* px = x + ((size_t)b * CDIM) * LL + hg * WW + wg;
      float ss = 0.f;
      for (int c = 0; c < 64; ++c) { float v = px[(size_t)c * LL]; ss += v * v; }
      float inv = rsqrtf(ss / 64.f + EPSV);
      for (int c = 0; c < 64; ++c)
        xn[t * 64 + c] = (_Float16)(px[(size_t)c * LL] * inv);
    } else {
      for (int c = 0; c < 64; ++c) xn[t * 64 + c] = (_Float16)0.f;
    }
  }
  // zero pad rows of o (softmax only fills rows < 49)
  for (int i = tid; i < 15 * 64; i += 256) osm[49 * 64 + i] = (_Float16)0.f;
  __syncthreads();

  // QKV: 16 tiles (4x4), 2 per wave, K=64; zero A-rows give zero C-rows
  for (int t = wave; t < 16; t += 8) {
    int m0 = (t >> 2) * 16, n0 = (t & 3) * 16;
    v8f cq = {}; v8f ck = {}; v8f cv = {};
#pragma unroll
    for (int kk = 0; kk < 64; kk += 32) {
      v16h a = frag_A_lds(xn, 64, m0, kk, lane);
      cq = wmma_f16(a, frag_B_WT_g(wq, 64, n0, kk, lane, gq), cq);
      ck = wmma_f16(a, frag_B_WT_g(wk, 64, n0, kk, lane, gk), ck);
      cv = wmma_f16(a, frag_B_WT_g(wv, 64, n0, kk, lane, gv), cv);
    }
    int n = n0 + (lane & 15), mb = (lane >= 16) ? 8 : 0;
#pragma unroll
    for (int r = 0; r < 8; ++r) {
      int m = m0 + r + mb;
      qs[m * 64 + n] = (_Float16)cq[r];
      ksm[m * 64 + n] = (_Float16)ck[r];
      vsm[m * 64 + n] = (_Float16)cv[r];
    }
  }
  __syncthreads();

  // online-softmax attention, hd=8, fp32, vector (b128) LDS reads
  for (int task = tid; task < NHEAD * 49; task += 256) {
    int hh = task / 49, row = task % 49, co = hh * 8;
    v8h qv = *(const v8h*)(qs + row * 64 + co);
    float qr[8];
#pragma unroll
    for (int d = 0; d < 8; ++d) qr[d] = (float)qv[d];
    float mx = -1e30f, se = 0.f;
    float acc[8] = {0, 0, 0, 0, 0, 0, 0, 0};
    for (int j = 0; j < 49; ++j) {
      v8h kv = *(const v8h*)(ksm + j * 64 + co);
      v8h vv = *(const v8h*)(vsm + j * 64 + co);
      float s = 0.f;
#pragma unroll
      for (int d = 0; d < 8; ++d) s += qr[d] * (float)kv[d];
      s *= 0.35355339059f;  // 8^-0.5
      float nm = fmaxf(mx, s);
      float corr = expf(mx - nm), p = expf(s - nm);
      se = se * corr + p;
#pragma unroll
      for (int d = 0; d < 8; ++d) acc[d] = acc[d] * corr + p * (float)vv[d];
      mx = nm;
    }
    float inv = 1.0f / se;
    v8h ov;
#pragma unroll
    for (int d = 0; d < 8; ++d) ov[d] = (_Float16)(acc[d] * inv);
    *(v8h*)(osm + row * 64 + co) = ov;
  }
  __syncthreads();

  // proj + window_reverse scatter (y channels 0..63)
  for (int t = wave; t < 16; t += 8) {
    int m0 = (t >> 2) * 16, n0 = (t & 3) * 16;
    v8f c = {};
#pragma unroll
    for (int kk = 0; kk < 64; kk += 32)
      c = wmma_f16(frag_A_lds(osm, 64, m0, kk, lane),
                   frag_B_WT(wproj, 64, n0, kk, lane), c);
    int n = n0 + (lane & 15), mb = (lane >= 16) ? 8 : 0;
    float bb = bproj[n];
#pragma unroll
    for (int r = 0; r < 8; ++r) {
      int m = m0 + r + mb;
      if (m < 49) {
        int hg = wh * 7 + m / 7, wg = ww * 7 + m % 7;
        y[((size_t)b * CDIM + n) * LL + hg * WW + wg] = c[r] + bb;
      }
    }
  }
}

// ======================= 2) mamba in-proj =======================
__global__ __launch_bounds__(256)
void k_inproj(const float* __restrict__ x, const float* __restrict__ gvm,
              const float* __restrict__ m_in_w, float* __restrict__ xin) {
  __shared__ __align__(16) _Float16 A[64 * 64];
  int tid = threadIdx.x, lane = tid & 31, wave = tid >> 5;
  int b = blockIdx.x / 49, l0 = (blockIdx.x % 49) * 64;
  for (int m = tid; m < 64; m += 256) {
    const float* px = x + ((size_t)b * CDIM + 64) * LL + l0 + m;
    float ss = 0.f;
    for (int c = 0; c < 64; ++c) { float v = px[(size_t)c * LL]; ss += v * v; }
    float inv = rsqrtf(ss / 64.f + EPSV);
    for (int c = 0; c < 64; ++c)
      A[m * 64 + c] = (_Float16)(px[(size_t)c * LL] * inv * gvm[c]);
  }
  __syncthreads();
  for (int t = wave; t < 32; t += 8) {  // M4 x N8
    int m0 = (t >> 3) * 16, n0 = (t & 7) * 16;
    v8f c = {};
#pragma unroll
    for (int kk = 0; kk < 64; kk += 32)
      c = wmma_f16(frag_A_lds(A, 64, m0, kk, lane),
                   frag_B_WT(m_in_w, 64, n0, kk, lane), c);
    int n = n0 + (lane & 15), mb = (lane >= 16) ? 8 : 0;
    // 8 consecutive pixels per lane -> two b128 stores
    float* dst = xin + ((size_t)b * DIc + n) * LL + l0 + m0 + mb;
    v4f s0 = {c[0], c[1], c[2], c[3]};
    v4f s1 = {c[4], c[5], c[6], c[7]};
    *(v4f*)dst = s0;
    *(v4f*)(dst + 4) = s1;
  }
}

// ======================= 3) depthwise 3x3 + bias + silu =======================
__global__ __launch_bounds__(256)
void k_dwconv(const float* __restrict__ xin, const float* __restrict__ cw,
              const float* __restrict__ cb, float* __restrict__ xconv) {
  size_t idx = (size_t)blockIdx.x * 256 + threadIdx.x;
  int wgl = idx % WW;
  int hgl = (idx / WW) % HH;
  int d = (idx / LL) % DIc;
  int b = idx / ((size_t)DIc * LL);
  const float* src = xin + ((size_t)b * DIc + d) * LL;
  const float* w = cw + d * 9;
  float s = 0.f;
#pragma unroll
  for (int dy = -1; dy <= 1; ++dy)
#pragma unroll
    for (int dx = -1; dx <= 1; ++dx) {
      int h2 = hgl + dy, w2 = wgl + dx;
      if (h2 >= 0 && h2 < HH && w2 >= 0 && w2 < WW)
        s += w[(dy + 1) * 3 + (dx + 1)] * src[h2 * WW + w2];
    }
  s += cb[d];
  xconv[idx] = s * sigm(s);
}

// ======================= 4) x_dbl projection + dt/softplus =======================
__global__ __launch_bounds__(256)
void k_xdbl(const float* __restrict__ xconv, const float* __restrict__ m_xproj,
            const float* __restrict__ m_dtw, const float* __restrict__ m_dtb,
            float* __restrict__ delta_g, float* __restrict__ Bs_g,
            float* __restrict__ Cs_g) {
  __shared__ __align__(16) _Float16 A[64 * 128];
  __shared__ __align__(16) _Float16 WT[48 * 128];  // xproj transposed, padded 36->48
  __shared__ float xd[64 * 48];
  int tid = threadIdx.x, lane = tid & 31, wave = tid >> 5;
  int bk = blockIdx.x / 49, l0 = (blockIdx.x % 49) * 64;
  int b = bk >> 2, k = bk & 3;
  const float* Wp = m_xproj + (size_t)k * DIc * 36;
  for (int i = tid; i < 64 * 128; i += 256) {
    int m = i >> 7, d = i & 127;
    A[i] = (_Float16)xconv[((size_t)b * DIc + d) * LL + scan_src(k, l0 + m)];
  }
  for (int i = tid; i < 48 * 128; i += 256) {
    int c = i >> 7, d = i & 127;
    WT[i] = (c < 36) ? (_Float16)Wp[d * 36 + c] : (_Float16)0.f;
  }
  __syncthreads();
  for (int t = wave; t < 12; t += 8) {  // M4 x N3 (48 cols, pad is zero weight)
    int m0 = (t / 3) * 16, n0 = (t % 3) * 16;
    v8f c = {};
#pragma unroll
    for (int kk = 0; kk < 128; kk += 32)
      c = wmma_f16(frag_A_lds(A, 128, m0, kk, lane),
                   frag_A_lds(WT, 128, n0, kk, lane), c);  // B from LDS [n][k]
    int n = n0 + (lane & 15), mb = (lane >= 16) ? 8 : 0;
#pragma unroll
    for (int r = 0; r < 8; ++r) xd[(m0 + r + mb) * 48 + n] = c[r];
  }
  __syncthreads();
  for (int i = tid; i < 64 * 16; i += 256) {
    int m = i >> 4, n = i & 15;
    size_t base = ((size_t)bk * LL + l0 + m) * DSTn + n;
    Bs_g[base] = xd[m * 48 + 4 + n];
    Cs_g[base] = xd[m * 48 + 20 + n];
  }
  for (int i = tid; i < 64 * 128; i += 256) {
    int m = i >> 7, d = i & 127;
    const float* dw = m_dtw + ((size_t)k * DIc + d) * DTRn;
    float dt = xd[m * 48 + 0] * dw[0] + xd[m * 48 + 1] * dw[1] +
               xd[m * 48 + 2] * dw[2] + xd[m * 48 + 3] * dw[3] + m_dtb[k * DIc + d];
    float sp = (dt > 20.f) ? dt : log1pf(expf(dt));
    delta_g[((size_t)bk * LL + l0 + m) * DIc + d] = sp;
  }
}

// ======================= 5) selective scan =======================
#define SCH 16
__global__ __launch_bounds__(128)
void k_scan(const float* __restrict__ xconv, const float* __restrict__ delta_g,
            const float* __restrict__ Bs_g, const float* __restrict__ Cs_g,
            const float* __restrict__ m_Alog, const float* __restrict__ m_D,
            float* __restrict__ ys_g) {
  __shared__ float Bsh[SCH * 16];
  __shared__ float Csh[SCH * 16];
  int d = threadIdx.x;
  int bk = blockIdx.x, b = bk >> 2, k = bk & 3;
  float Arow[16], h[16];
#pragma unroll
  for (int n = 0; n < 16; ++n) {
    Arow[n] = -expf(m_Alog[((size_t)k * DIc + d) * DSTn + n]);
    h[n] = 0.f;
  }
  float Dv = m_D[k * DIc + d];
  const float* xc = xconv + ((size_t)b * DIc + d) * LL;
  size_t base = (size_t)bk * LL;
  for (int l0 = 0; l0 < LL; l0 += SCH) {
    for (int i = threadIdx.x; i < SCH * 16; i += 128) {
      Bsh[i] = Bs_g[(base + l0 + (i >> 4)) * DSTn + (i & 15)];
      Csh[i] = Cs_g[(base + l0 + (i >> 4)) * DSTn + (i & 15)];
    }
    __syncthreads();
    if (l0 + SCH < LL)  // global_prefetch next delta chunk
      __builtin_prefetch(&delta_g[(base + l0 + SCH) * DIc + d], 0, 1);
    for (int j = 0; j < SCH; ++j) {
      int l = l0 + j;
      float delta = delta_g[(base + l) * DIc + d];
      float u = xc[scan_src(k, l)];
      float du = delta * u, yv = 0.f;
#pragma unroll
      for (int n = 0; n < 16; ++n) {
        float e = __expf(delta * Arow[n]);
        h[n] = e * h[n] + du * Bsh[j * 16 + n];
        yv += h[n] * Csh[j * 16 + n];
      }
      ys_g[(base + l) * DIc + d] = yv + u * Dv;
    }
    __syncthreads();
  }
}

// ======================= 6) cross-merge + gelu + out-proj =======================
__global__ __launch_bounds__(256)
void k_merge(const float* __restrict__ ys_g, const float* __restrict__ m_out_w,
             float* __restrict__ y) {
  __shared__ __align__(16) _Float16 A[64 * 128];
  int tid = threadIdx.x, lane = tid & 31, wave = tid >> 5;
  int b = blockIdx.x / 49, l0 = (blockIdx.x % 49) * 64;
  size_t base = (size_t)b * 4 * LL;
  for (int i = tid; i < 64 * 128; i += 256) {
    int m = i >> 7, d = i & 127;
    int l = l0 + m;
    int lT = (l % WW) * WW + (l / WW);
    float v = ys_g[(base + l) * DIc + d] +
              ys_g[(base + 2 * LL + (LL - 1 - l)) * DIc + d] +
              ys_g[(base + LL + lT) * DIc + d] +
              ys_g[(base + 3 * LL + (LL - 1 - lT)) * DIc + d];
    A[i] = (_Float16)gelu_t(v);
  }
  __syncthreads();
  for (int t = wave; t < 16; t += 8) {  // M4 x N4, K=128
    int m0 = (t >> 2) * 16, n0 = (t & 3) * 16;
    v8f c = {};
#pragma unroll
    for (int kk = 0; kk < 128; kk += 32)
      c = wmma_f16(frag_A_lds(A, 128, m0, kk, lane),
                   frag_B_WT(m_out_w, 128, n0, kk, lane), c);
    int n = n0 + (lane & 15), mb = (lane >= 16) ? 8 : 0;
    float* dst = y + ((size_t)b * CDIM + 64 + n) * LL + l0 + m0 + mb;
    v4f s0 = {c[0], c[1], c[2], c[3]};
    v4f s1 = {c[4], c[5], c[6], c[7]};
    *(v4f*)dst = s0;
    *(v4f*)(dst + 4) = s1;
  }
}

// ======================= 7) SE =======================
__global__ __launch_bounds__(256)
void k_pool(const float* __restrict__ y, float* __restrict__ p) {
  __shared__ float red[256];
  int bc = blockIdx.x;
  const float* src = y + (size_t)bc * LL;
  float s = 0.f;
  for (int i = threadIdx.x; i < LL; i += 256) s += src[i];
  red[threadIdx.x] = s;
  __syncthreads();
  for (int st = 128; st > 0; st >>= 1) {
    if (threadIdx.x < st) red[threadIdx.x] += red[threadIdx.x + st];
    __syncthreads();
  }
  if (threadIdx.x == 0) p[bc] = red[0] / (float)LL;
}

__global__ __launch_bounds__(128)
void k_se(const float* __restrict__ p, const float* __restrict__ w1,
          const float* __restrict__ b1, const float* __restrict__ w2,
          const float* __restrict__ b2, float* __restrict__ s) {
  __shared__ float t[32];
  int b = blockIdx.x;
  const float* pb = p + b * CDIM;
  if (threadIdx.x < 32) {
    float a = b1[threadIdx.x];
    for (int c = 0; c < CDIM; ++c) a += pb[c] * w1[threadIdx.x * CDIM + c];
    t[threadIdx.x] = fmaxf(a, 0.f);
  }
  __syncthreads();
  int c = threadIdx.x;
  float a = b2[c];
#pragma unroll
  for (int i = 0; i < 32; ++i) a += t[i] * w2[c * 32 + i];
  s[b * CDIM + c] = sigm(a);
}

__global__ __launch_bounds__(256)
void k_res1(const float* __restrict__ x, const float* __restrict__ y,
            const float* __restrict__ s, const float* __restrict__ g1,
            float* __restrict__ xres) {
  size_t idx = (size_t)blockIdx.x * 256 + threadIdx.x;
  int c = (idx / LL) % CDIM;
  int b = idx / ((size_t)CDIM * LL);
  xres[idx] = x[idx] + g1[c] * y[idx] * s[b * CDIM + c];
}

// ======================= 8) fused MLP =======================
__global__ __launch_bounds__(256)
void k_mlp(const float* __restrict__ xres, const float* __restrict__ gmlp,
           const float* __restrict__ w1, const float* __restrict__ b1,
           const float* __restrict__ w2, const float* __restrict__ b2,
           const float* __restrict__ g2, float* __restrict__ out) {
  __shared__ __align__(16) _Float16 A[32 * 128];
  __shared__ __align__(16) _Float16 Hl[32 * 512];
  int tid = threadIdx.x, lane = tid & 31, wave = tid >> 5;
  int b = blockIdx.x / 98, l0 = (blockIdx.x % 98) * 32;
  for (int m = tid; m < 32; m += 256) {
    const float* px = xres + ((size_t)b * CDIM) * LL + l0 + m;
    float ss = 0.f;
    for (int c = 0; c < CDIM; ++c) { float v = px[(size_t)c * LL]; ss += v * v; }
    float inv = rsqrtf(ss / 128.f + EPSV);
    for (int c = 0; c < CDIM; ++c)
      A[m * 128 + c] = (_Float16)(px[(size_t)c * LL] * inv * gmlp[c]);
  }
  __syncthreads();
  // GEMM1: 32x512, M2 x N32, K=128
  for (int t = wave; t < 64; t += 8) {
    int m0 = (t >> 5) * 16, n0 = (t & 31) * 16;
    v8f c = {};
#pragma unroll
    for (int kk = 0; kk < 128; kk += 32)
      c = wmma_f16(frag_A_lds(A, 128, m0, kk, lane),
                   frag_B_WT(w1, 128, n0, kk, lane), c);
    int n = n0 + (lane & 15), mb = (lane >= 16) ? 8 : 0;
    float bb = b1[n];
#pragma unroll
    for (int r = 0; r < 8; ++r)
      Hl[(m0 + r + mb) * 512 + n] = (_Float16)gelu_t(c[r] + bb);
  }
  __syncthreads();
  // GEMM2: 32x128, M2 x N8, K=512; vectorized residual epilogue
  for (int t = wave; t < 16; t += 8) {
    int m0 = (t >> 3) * 16, n0 = (t & 7) * 16;
    v8f c = {};
#pragma unroll
    for (int kk = 0; kk < 512; kk += 32)
      c = wmma_f16(frag_A_lds(Hl, 512, m0, kk, lane),
                   frag_B_WT(w2, 512, n0, kk, lane), c);
    int n = n0 + (lane & 15), mb = (lane >= 16) ? 8 : 0;
    float bb = b2[n], gg = g2[n];
    size_t off = ((size_t)b * CDIM + n) * LL + l0 + m0 + mb;
    const float* pr = xres + off;
    v4f r0 = *(const v4f*)pr;
    v4f r1 = *(const v4f*)(pr + 4);
    v4f o0, o1;
#pragma unroll
    for (int i = 0; i < 4; ++i) {
      o0[i] = r0[i] + gg * (c[i] + bb);
      o1[i] = r1[i] + gg * (c[i + 4] + bb);
    }
    *(v4f*)(out + off) = o0;
    *(v4f*)(out + off + 4) = o1;
  }
}

// ======================= launch =======================
extern "C" void kernel_launch(void* const* d_in, const int* in_sizes, int n_in,
                              void* d_out, int out_size, void* d_ws, size_t ws_size,
                              hipStream_t stream) {
  (void)in_sizes; (void)n_in; (void)out_size; (void)ws_size;
  const float* x       = (const float*)d_in[0];
  const float* wq      = (const float*)d_in[1];
  const float* wk      = (const float*)d_in[2];
  const float* wv      = (const float*)d_in[3];
  const float* wproj   = (const float*)d_in[4];
  const float* bproj   = (const float*)d_in[5];
  const float* gq      = (const float*)d_in[6];
  const float* gk      = (const float*)d_in[7];
  const float* gv      = (const float*)d_in[8];
  const float* gvm     = (const float*)d_in[9];
  const float* gmlp    = (const float*)d_in[10];
  const float* m_in_w  = (const float*)d_in[11];
  const float* m_conv_w= (const float*)d_in[12];
  const float* m_conv_b= (const float*)d_in[13];
  const float* m_xproj = (const float*)d_in[14];
  const float* m_dtw   = (const float*)d_in[15];
  const float* m_dtb   = (const float*)d_in[16];
  const float* m_Alog  = (const float*)d_in[17];
  const float* m_D     = (const float*)d_in[18];
  const float* m_out_w = (const float*)d_in[19];
  const float* se_w1   = (const float*)d_in[20];
  const float* se_b1   = (const float*)d_in[21];
  const float* se_w2   = (const float*)d_in[22];
  const float* se_b2   = (const float*)d_in[23];
  const float* mlp_w1  = (const float*)d_in[24];
  const float* mlp_b1  = (const float*)d_in[25];
  const float* mlp_w2  = (const float*)d_in[26];
  const float* mlp_b2  = (const float*)d_in[27];
  const float* gamma1  = (const float*)d_in[28];
  const float* gamma2  = (const float*)d_in[29];

  float* ws = (float*)d_ws;
  size_t o = 0;
  float* y     = ws + o; o += (size_t)NB * CDIM * LL;       // y_attn | y_mamba
  float* xin   = ws + o; o += (size_t)NB * DIc * LL;
  float* xconv = ws + o; o += (size_t)NB * DIc * LL;
  float* delta = ws + o; o += (size_t)NB * KDIR * LL * DIc;
  float* Bsb   = ws + o; o += (size_t)NB * KDIR * LL * DSTn;
  float* Csb   = ws + o; o += (size_t)NB * KDIR * LL * DSTn;
  float* ysb   = ws + o; o += (size_t)NB * KDIR * LL * DIc;
  float* pbuf  = ws + o; o += (size_t)NB * CDIM;
  float* sbuf  = ws + o; o += (size_t)NB * CDIM;
  float* xres  = ws + o; o += (size_t)NB * CDIM * LL;

  k_attn  <<<NB * 64, 256, 0, stream>>>(x, wq, wk, wv, wproj, bproj, gq, gk, gv, y);
  k_inproj<<<NB * 49, 256, 0, stream>>>(x, gvm, m_in_w, xin);
  k_dwconv<<<(NB * DIc * LL) / 256, 256, 0, stream>>>(xin, m_conv_w, m_conv_b, xconv);
  k_xdbl  <<<NB * KDIR * 49, 256, 0, stream>>>(xconv, m_xproj, m_dtw, m_dtb,
                                               delta, Bsb, Csb);
  k_scan  <<<NB * KDIR, 128, 0, stream>>>(xconv, delta, Bsb, Csb, m_Alog, m_D, ysb);
  k_merge <<<NB * 49, 256, 0, stream>>>(ysb, m_out_w, y);
  k_pool  <<<NB * CDIM, 256, 0, stream>>>(y, pbuf);
  k_se    <<<NB, 128, 0, stream>>>(pbuf, se_w1, se_b1, se_w2, se_b2, sbuf);
  k_res1  <<<(NB * CDIM * LL) / 256, 256, 0, stream>>>(x, y, sbuf, gamma1, xres);
  k_mlp   <<<NB * 98, 256, 0, stream>>>(xres, gmlp, mlp_w1, mlp_b1, mlp_w2, mlp_b2,
                                        gamma2, (float*)d_out);
}
